// GeometryGatedCrossAttention_64733747085443
// MI455X (gfx1250) — matrix operand order repaired and checked by reading, not compiled, not run on hardware
//
#include <hip/hip_runtime.h>
#include <cmath>

typedef _Float16 half_t;
typedef __attribute__((ext_vector_type(16))) _Float16 v16h;
typedef __attribute__((ext_vector_type(8)))  _Float16 v8h;
typedef __attribute__((ext_vector_type(8)))  float    v8f;

// ---------------- problem constants ----------------
constexpr int kB    = 8;
constexpr int kP    = 16384;
constexpr int kS    = 77;
constexpr int kSP   = 80;   // S padded to 16 multiple for score tiles
constexpr int kSPAD = 96;   // S padded to 32 multiple for attn@V k-dim
constexpr int kDIM  = 64;
constexpr int kINNER= 256;
constexpr int kH    = 8;
constexpr int kDH   = 32;
constexpr int kCTX  = 768;
constexpr int kFFN  = 1024;
constexpr long kN   = (long)kB * kP;     // 131072

// ---------------- WMMA fragment helpers ----------------
// A fragment (16x32, f16, row-major source, lda in halves):
//   lanes 0-15 : row M=lane,    K = {0..7, 16..23}
//   lanes 16-31: row M=lane-16, K = {8..15, 24..31}
union AFrag { v16h v; v8h h[2]; };

__device__ inline v16h load_a_frag(const half_t* A, int lda, int kbase) {
  const int lane = threadIdx.x & 31;
  const int r    = lane & 15;
  const int koff = (lane >> 4) << 3;           // 0 or 8
  const half_t* p = A + (size_t)r * lda + kbase + koff;
  AFrag f;
  f.h[0] = *(const v8h*)(p);
  f.h[1] = *(const v8h*)(p + 16);
  return f.v;
}

// B fragment (32x16, f16) from transposed weights Wt[n][k], row stride bstride.
//   lanes 0-15 : col N=n0+lane,    K = kbase+{0..7, 16..23}
//   lanes 16-31: col N=n0+lane-16, K = kbase+{8..15, 24..31}
__device__ inline v16h load_b_frag(const half_t* Wt, int n0, int bstride, int kbase) {
  const int lane = threadIdx.x & 31;
  const int n    = n0 + (lane & 15);
  const int koff = (lane >> 4) << 3;           // 0 or 8
  const half_t* p = Wt + (size_t)n * bstride + kbase + koff;
  AFrag f;
  f.h[0] = *(const v8h*)(p);
  f.h[1] = *(const v8h*)(p + 16);
  return f.v;
}

__device__ inline v8f wmma_f16(v16h a, v16h b, v8f c) {
  return __builtin_amdgcn_wmma_f32_16x16x32_f16(false, a, false, b, (short)0, c,
                                                false, false);
}

// ---------------- generic WMMA GEMM with LDS-staged weights ----------------
// C[M,Nt] = A[M,K] (f16) * Wt[Nt,K]^T (f16) + bias, epilogue by mode.
// block = 256 thr, 8 waves x 32 rows = 256 rows, grid.y covers 64-col groups.
// The 64xK weight tile for the block is staged once in dynamic LDS; each B
// fragment read from LDS feeds two WMMAs (two 16-row A fragments per wave);
// the A stream is register double-buffered so global latency overlaps WMMAs.
// mode: 0 = store f32, 1 = store f16, 2 = exact-GELU -> f16, 3 = += into f32
__global__ __launch_bounds__(256) void gemm_f16_wmma(
    const half_t* __restrict__ A, int lda,
    const half_t* __restrict__ Wt, int bstride,
    const float* __restrict__ bias, int has_bias,
    void* __restrict__ Cp, int ldc, int K, int M, int mode) {
  extern __shared__ char smem[];
  half_t* Wlds = (half_t*)smem;                // [64][K]

  const int wave = threadIdx.x >> 5;
  const int lane = threadIdx.x & 31;
  const int row0 = blockIdx.x * 256 + wave * 32;
  const int col0 = blockIdx.y * 64;

  // cooperative copy of the 64-column weight tile into LDS (16B chunks)
  {
    const int chunks_per_row = K >> 3;         // v8h chunks per row
    const int total = 64 * chunks_per_row;
    for (int i = threadIdx.x; i < total; i += 256) {
      const int r  = i / chunks_per_row;
      const int cc = i - r * chunks_per_row;
      *(v8h*)(Wlds + (size_t)r * K + cc * 8) =
          *(const v8h*)(Wt + (size_t)(col0 + r) * bstride + cc * 8);
    }
  }
  __syncthreads();

  if (row0 >= M) return;                       // wave-uniform tail guard

  const half_t* Arow0 = A + (size_t)row0 * lda;
  const half_t* Arow1 = Arow0 + (size_t)16 * lda;

  v8f acc[8] = {};                             // [2 row frags][4 col tiles]

  v16h a0 = load_a_frag(Arow0, lda, 0);        // prologue
  v16h a1 = load_a_frag(Arow1, lda, 0);
  for (int kb = 0; ; kb += 32) {
    v16h c0 = a0, c1 = a1;
    const bool more = (kb + 32 < K);
    if (more) {                                // pipeline next A fragments
      __builtin_prefetch((const void*)(Arow0 + (size_t)(lane & 15) * lda + kb + 64), 0, 3);
      a0 = load_a_frag(Arow0, lda, kb + 32);
      a1 = load_a_frag(Arow1, lda, kb + 32);
    }
    #pragma unroll
    for (int t = 0; t < 4; ++t) {
      v16h b = load_b_frag(Wlds, t * 16, K, kb);
      acc[t]     = wmma_f16(c0, b, acc[t]);
      acc[4 + t] = wmma_f16(c1, b, acc[4 + t]);
    }
    if (!more) break;
  }

  // D layout: VGPR i -> row = rowbase + 8*(lane>=16) + i, col = col tile + (lane&15)
  const int cl = lane & 15;
  const int rhalf = (lane >> 4) << 3;
  #pragma unroll
  for (int f = 0; f < 2; ++f) {
    const int rb = row0 + f * 16 + rhalf;
    #pragma unroll
    for (int t = 0; t < 4; ++t) {
      const int col = col0 + t * 16 + cl;
      const float bv = has_bias ? bias[col] : 0.0f;
      v8f av = acc[f * 4 + t];
      #pragma unroll
      for (int i = 0; i < 8; ++i) {
        float v = av[i] + bv;
        size_t idx = (size_t)(rb + i) * ldc + col;
        if (mode == 0) {
          ((float*)Cp)[idx] = v;
        } else if (mode == 1) {
          ((half_t*)Cp)[idx] = (half_t)v;
        } else if (mode == 2) {
          float g = 0.5f * v * (1.0f + erff(v * 0.70710678118654752f));
          ((half_t*)Cp)[idx] = (half_t)g;
        } else {
          ((float*)Cp)[idx] += v;
        }
      }
    }
  }
}

// ---------------- LayerNorm (row=256) f32 -> f16 ----------------
__global__ __launch_bounds__(256) void ln_f32_to_f16(
    const float* __restrict__ X, const float* __restrict__ gam,
    const float* __restrict__ bet, half_t* __restrict__ Y) {
  const int wave = threadIdx.x >> 5;
  const int lane = threadIdx.x & 31;
  const size_t row = (size_t)blockIdx.x * 8 + wave;
  const float* x = X + row * kINNER;
  float v[8], s = 0.f, s2 = 0.f;
  #pragma unroll
  for (int i = 0; i < 8; ++i) { v[i] = x[lane * 8 + i]; s += v[i]; s2 += v[i] * v[i]; }
  #pragma unroll
  for (int m = 16; m >= 1; m >>= 1) { s += __shfl_xor(s, m, 32); s2 += __shfl_xor(s2, m, 32); }
  const float mean = s * (1.0f / kINNER);
  const float var  = s2 * (1.0f / kINNER) - mean * mean;
  const float rinv = rsqrtf(var + 1e-5f);
  half_t* y = Y + row * kINNER;
  #pragma unroll
  for (int i = 0; i < 8; ++i) {
    int c = lane * 8 + i;
    y[c] = (half_t)((v[i] - mean) * rinv * gam[c] + bet[c]);
  }
}

// ---------------- small utility kernels ----------------
__global__ void cast_f32_f16(const float* __restrict__ X, half_t* __restrict__ Y, long n) {
  long i = (long)blockIdx.x * 256 + threadIdx.x;
  if (i < n) Y[i] = (half_t)X[i];
}

// W[R][C] f32 -> Wt[C][R] f16
__global__ void transpose_to_f16(const float* __restrict__ W, half_t* __restrict__ Wt,
                                 int R, int C) {
  int i = blockIdx.x * 256 + threadIdx.x;
  if (i >= R * C) return;
  int r = i / C, c = i % C;
  Wt[(size_t)c * R + r] = (half_t)W[i];
}

// context [B][77][768] f32 -> [B][80][768] f16, zero-padded rows
__global__ void ctx_pad_f16(const float* __restrict__ ctx, half_t* __restrict__ o) {
  int i = blockIdx.x * 256 + threadIdx.x;
  const int tot = kB * kSP * kCTX;
  if (i >= tot) return;
  int c = i % kCTX; int t = i / kCTX; int s = t % kSP; int b = t / kSP;
  o[i] = (s < kS) ? (half_t)ctx[((size_t)b * kS + s) * kCTX + c] : (half_t)0.0f;
}

// kvtmp [B*80][256] -> Kbuf [b][h][80][32] (zero for s>=77)
__global__ void reshape_k(const half_t* __restrict__ kt, half_t* __restrict__ Kbuf) {
  int i = blockIdx.x * 256 + threadIdx.x;
  const int tot = kB * kH * kSP * kDH;
  if (i >= tot) return;
  int d = i & 31; int t = i >> 5; int s = t % kSP; t /= kSP; int h = t & 7; int b = t >> 3;
  Kbuf[i] = (s < kS) ? kt[((size_t)b * kSP + s) * kINNER + h * kDH + d] : (half_t)0.0f;
}

// kvtmp -> Vt [b][h][32][96] (V transposed, zero for s>=77 and pad region)
__global__ void reshape_vt(const half_t* __restrict__ vt, half_t* __restrict__ Vt) {
  int i = blockIdx.x * 256 + threadIdx.x;
  const int tot = kB * kH * kDH * kSPAD;
  if (i >= tot) return;
  int s = i % kSPAD; int t = i / kSPAD; int d = t & 31; t >>= 5; int h = t & 7; int b = t >> 3;
  Vt[i] = (s < kS) ? vt[((size_t)b * kSP + s) * kINNER + h * kDH + d] : (half_t)0.0f;
}

// ---------------- fused attention per (b,h), 64 points / block ----------------
__global__ __launch_bounds__(128) void attn_wmma(
    const half_t* __restrict__ Q,     // [N][256] f16
    const half_t* __restrict__ Kbuf,  // [b][h][80][32]
    const half_t* __restrict__ Vt,    // [b][h][32][96]
    half_t* __restrict__ O) {         // [N][256] f16
  __shared__ float  sc[4][16][kSPAD];
  __shared__ half_t pr[4][16][kSPAD];
  const int w = threadIdx.x >> 5, lane = threadIdx.x & 31;
  const int bh = blockIdx.x;
  const int b = bh >> 3, h = bh & 7;
  const int p0 = blockIdx.y * 64 + w * 16;

  const half_t* Qrow = Q + ((size_t)b * kP + p0) * kINNER + h * kDH;
  v16h aq = load_a_frag(Qrow, kINNER, 0);
  const half_t* Kbh = Kbuf + (size_t)bh * kSP * kDH;
  const float scale = 0.17677669529663687f;   // 1/sqrt(32)

  #pragma unroll
  for (int st = 0; st < 5; ++st) {            // 5 key tiles of 16
    v8f acc = {};
    v16h bk = load_b_frag(Kbh, st * 16, kDH, 0);
    acc = wmma_f16(aq, bk, acc);
    const int cl = lane & 15, rbase = (lane >> 4) << 3;
    const int s = st * 16 + cl;
    #pragma unroll
    for (int i = 0; i < 8; ++i)
      sc[w][rbase + i][s] = (s < kS) ? acc[i] * scale : -1e30f;
  }
  __syncthreads();

  if (lane < 16) {                            // softmax over 80, one row per lane
    float m = -1e30f;
    for (int j = 0; j < kSP; ++j) m = fmaxf(m, sc[w][lane][j]);
    float sum = 0.f;
    for (int j = 0; j < kSP; ++j) { float e = expf(sc[w][lane][j] - m); sum += e; sc[w][lane][j] = e; }
    float inv = 1.0f / sum;
    for (int j = 0; j < kSP; ++j)   pr[w][lane][j] = (half_t)(sc[w][lane][j] * inv);
    for (int j = kSP; j < kSPAD; ++j) pr[w][lane][j] = (half_t)0.0f;
  }
  __syncthreads();

  const half_t* Vbh = Vt + (size_t)bh * kDH * kSPAD;
  const half_t* Ap  = &pr[w][0][0];
  half_t* Orow = O + ((size_t)b * kP + p0) * kINNER;
  #pragma unroll
  for (int nt = 0; nt < 2; ++nt) {            // DH=32 -> 2 col tiles
    v8f acc = {};
    #pragma unroll
    for (int kc = 0; kc < 3; ++kc) {          // 96 = 3 k chunks
      v16h a  = load_a_frag(Ap, kSPAD, kc * 32);
      v16h bv = load_b_frag(Vbh, nt * 16, kSPAD, kc * 32);
      acc = wmma_f16(a, bv, acc);
    }
    const int cl = lane & 15, rbase = (lane >> 4) << 3;
    const int col = h * kDH + nt * 16 + cl;
    #pragma unroll
    for (int i = 0; i < 8; ++i)
      Orow[(size_t)(rbase + i) * kINNER + col] = (half_t)acc[i];
  }
}

// ---------------- geometry gate MLP + final residual (fp32 scalar) ----------------
__global__ __launch_bounds__(256) void gate_final(
    const float* __restrict__ data, const float* __restrict__ normals,
    const float* __restrict__ hout,
    const float* __restrict__ g1w, const float* __restrict__ g1b,
    const float* __restrict__ g2w, const float* __restrict__ g2b,
    const float* __restrict__ g3w, const float* __restrict__ g3b,
    const float* __restrict__ oscale, float* __restrict__ out) {
  long p = (long)blockIdx.x * 256 + threadIdx.x;
  if (p >= kN) return;
  float gin[67];
  const float* d = data + p * kDIM;
  #pragma unroll
  for (int i = 0; i < kDIM; ++i) gin[i] = d[i];
  float nx = normals[p * 3 + 0], ny = normals[p * 3 + 1], nz = normals[p * 3 + 2];
  float nn = sqrtf(nx * nx + ny * ny + nz * nz);
  float inv = 1.0f / fmaxf(nn, 1e-6f);
  gin[64] = nx * inv; gin[65] = ny * inv; gin[66] = nz * inv;

  float a1[32];
  for (int j = 0; j < 32; ++j) {
    float s = g1b[j];
    for (int i = 0; i < 67; ++i) s += gin[i] * g1w[i * 32 + j];
    a1[j] = s / (1.0f + expf(-s));            // silu
  }
  float a2[16];
  for (int k2 = 0; k2 < 16; ++k2) {
    float s = g2b[k2];
    for (int j = 0; j < 32; ++j) s += a1[j] * g2w[j * 16 + k2];
    a2[k2] = s / (1.0f + expf(-s));
  }
  float z = g3b[0];
  for (int k2 = 0; k2 < 16; ++k2) z += a2[k2] * g3w[k2];
  float gate = 1.0f / (1.0f + expf(-z));
  float sc = oscale[0] * gate;

  const float* hp = hout + p * kDIM;
  float* op = out + p * kDIM;
  #pragma unroll
  for (int i = 0; i < kDIM; ++i) op[i] = d[i] + sc * hp[i];
}

// ---------------- host orchestration ----------------
extern "C" void kernel_launch(void* const* d_in, const int* in_sizes, int n_in,
                              void* d_out, int out_size, void* d_ws, size_t ws_size,
                              hipStream_t stream) {
  const float* data       = (const float*)d_in[0];
  const float* context    = (const float*)d_in[1];
  const float* normals    = (const float*)d_in[2];
  const float* proj_in_w  = (const float*)d_in[3];
  const float* proj_in_b  = (const float*)d_in[4];
  const float* proj_out_w = (const float*)d_in[5];
  const float* proj_out_b = (const float*)d_in[6];
  const float* cn_g       = (const float*)d_in[7];
  const float* cn_b       = (const float*)d_in[8];
  const float* q_w        = (const float*)d_in[9];
  const float* q_b        = (const float*)d_in[10];
  const float* k_w        = (const float*)d_in[11];
  const float* k_b        = (const float*)d_in[12];
  const float* v_w        = (const float*)d_in[13];
  const float* v_b        = (const float*)d_in[14];
  const float* out_w      = (const float*)d_in[15];
  const float* out_b      = (const float*)d_in[16];
  const float* fn_g       = (const float*)d_in[17];
  const float* fn_b       = (const float*)d_in[18];
  const float* f1_w       = (const float*)d_in[19];
  const float* f1_b       = (const float*)d_in[20];
  const float* f2_w       = (const float*)d_in[21];
  const float* f2_b       = (const float*)d_in[22];
  const float* g1_w       = (const float*)d_in[23];
  const float* g1_b       = (const float*)d_in[24];
  const float* g2_w       = (const float*)d_in[25];
  const float* g2_b       = (const float*)d_in[26];
  const float* g3_w       = (const float*)d_in[27];
  const float* g3_b       = (const float*)d_in[28];
  const float* oscale     = (const float*)d_in[29];
  float* outp = (float*)d_out;

  char* ws = (char*)d_ws;
  size_t off = 0;
  auto alloc = [&](size_t bytes) -> void* {
    void* p = ws + off;
    off += (bytes + 255) & ~(size_t)255;
    return p;
  };

  half_t* wt_pi  = (half_t*)alloc((size_t)kINNER * kDIM * 2);     // [256][64]
  half_t* wt_q   = (half_t*)alloc((size_t)kINNER * kINNER * 2);   // [256][256]
  half_t* wt_out = (half_t*)alloc((size_t)kINNER * kINNER * 2);
  half_t* wt_k   = (half_t*)alloc((size_t)kINNER * kCTX * 2);     // [256][768]
  half_t* wt_v   = (half_t*)alloc((size_t)kINNER * kCTX * 2);
  half_t* wt_f1  = (half_t*)alloc((size_t)kFFN * kINNER * 2);     // [1024][256]
  half_t* wt_f2  = (half_t*)alloc((size_t)kINNER * kFFN * 2);     // [256][1024]
  half_t* wt_po  = (half_t*)alloc((size_t)kDIM * kINNER * 2);     // [64][256]
  half_t* data16 = (half_t*)alloc((size_t)kN * kDIM * 2);
  half_t* ctx16  = (half_t*)alloc((size_t)kB * kSP * kCTX * 2);
  half_t* kvtmp  = (half_t*)alloc((size_t)kB * kSP * kINNER * 2);
  half_t* Kbuf   = (half_t*)alloc((size_t)kB * kH * kSP * kDH * 2);
  half_t* Vt     = (half_t*)alloc((size_t)kB * kH * kDH * kSPAD * 2);
  float*  hbuf   = (float*) alloc((size_t)kN * kINNER * 4);
  half_t* X1     = (half_t*)alloc((size_t)kN * kINNER * 2);
  half_t* X2     = (half_t*)alloc((size_t)kN * kINNER * 2);
  float*  hout   = (float*) alloc((size_t)kN * kDIM * 4);

  auto T = [&](const float* W, half_t* Wt, int R, int C) {
    transpose_to_f16<<<(R * C + 255) / 256, 256, 0, stream>>>(W, Wt, R, C);
  };
  T(proj_in_w,  wt_pi,  kDIM,   kINNER);
  T(q_w,        wt_q,   kINNER, kINNER);
  T(out_w,      wt_out, kINNER, kINNER);
  T(k_w,        wt_k,   kCTX,   kINNER);
  T(v_w,        wt_v,   kCTX,   kINNER);
  T(f1_w,       wt_f1,  kINNER, kFFN);
  T(f2_w,       wt_f2,  kFFN,   kINNER);
  T(proj_out_w, wt_po,  kINNER, kDIM);

  cast_f32_f16<<<(int)((kN * kDIM + 255) / 256), 256, 0, stream>>>(data, data16, kN * kDIM);
  ctx_pad_f16<<<(kB * kSP * kCTX + 255) / 256, 256, 0, stream>>>(context, ctx16);

  const int kvM = kB * kSP;                              // 640 padded rows
  const dim3 gN4((int)(kN / 256), kINNER / 64);          // N rows, 256 cols
  const dim3 gN1((int)(kN / 256), 1);                    // N rows, 64 cols
  const dim3 gKV((kvM + 255) / 256, kINNER / 64);        // 640 rows -> 3 blocks
  auto shm = [](int K) { return (size_t)64 * K * 2; };   // LDS weight tile bytes

  // h = data @ proj_in_w + b     (f32)
  gemm_f16_wmma<<<gN4, 256, shm(kDIM), stream>>>(data16, kDIM, wt_pi, kDIM,
                                                 proj_in_b, 1, hbuf, kINNER, kDIM,
                                                 (int)kN, 0);
  // hn = LN(h)                    (f16 -> X1)
  ln_f32_to_f16<<<(int)(kN / 8), 256, 0, stream>>>(hbuf, cn_g, cn_b, X1);

  // K,V projections (640 padded rows)
  gemm_f16_wmma<<<gKV, 256, shm(kCTX), stream>>>(ctx16, kCTX, wt_k, kCTX, k_b, 1,
                                                 kvtmp, kINNER, kCTX, kvM, 1);
  reshape_k<<<(kB * kH * kSP * kDH + 255) / 256, 256, 0, stream>>>(kvtmp, Kbuf);
  gemm_f16_wmma<<<gKV, 256, shm(kCTX), stream>>>(ctx16, kCTX, wt_v, kCTX, v_b, 1,
                                                 kvtmp, kINNER, kCTX, kvM, 1);
  reshape_vt<<<(kB * kH * kDH * kSPAD + 255) / 256, 256, 0, stream>>>(kvtmp, Vt);

  // q = hn @ q_w + b              (f16 -> X2)
  gemm_f16_wmma<<<gN4, 256, shm(kINNER), stream>>>(X1, kINNER, wt_q, kINNER, q_b, 1,
                                                   X2, kINNER, kINNER, (int)kN, 1);
  // attention: o -> X1 (f16)
  attn_wmma<<<dim3(kB * kH, kP / 64), 128, 0, stream>>>(X2, Kbuf, Vt, X1);
  // h += o @ out_w + out_b
  gemm_f16_wmma<<<gN4, 256, shm(kINNER), stream>>>(X1, kINNER, wt_out, kINNER, out_b, 1,
                                                   hbuf, kINNER, kINNER, (int)kN, 3);
  // hf = LN(h) -> X2
  ln_f32_to_f16<<<(int)(kN / 8), 256, 0, stream>>>(hbuf, fn_g, fn_b, X2);

  // FFN in 4 column chunks of 256: X1 = GELU(hf @ f1 + b1); h += X1 @ f2 (+b2 once)
  for (int c = 0; c < 4; ++c) {
    gemm_f16_wmma<<<gN4, 256, shm(kINNER), stream>>>(
        X2, kINNER, wt_f1 + (size_t)c * 256 * kINNER, kINNER, f1_b + c * 256, 1,
        X1, kINNER, kINNER, (int)kN, 2);
    gemm_f16_wmma<<<gN4, 256, shm(kINNER), stream>>>(
        X1, kINNER, wt_f2 + (size_t)c * 256, kFFN, f2_b, (c == 0) ? 1 : 0,
        hbuf, kINNER, kINNER, (int)kN, 3);
  }

  // proj_out: hout = f16(h) @ proj_out_w + b (f32)
  cast_f32_f16<<<(int)((kN * kINNER + 255) / 256), 256, 0, stream>>>(hbuf, X2, kN * kINNER);
  gemm_f16_wmma<<<gN1, 256, shm(kINNER), stream>>>(X2, kINNER, wt_po, kINNER,
                                                   proj_out_b, 1, hout, kDIM, kINNER,
                                                   (int)kN, 0);

  // final: out = data + scale * gate * hout
  gate_final<<<(int)((kN + 255) / 256), 256, 0, stream>>>(
      data, normals, hout, g1_w, g1_b, g2_w, g2_b, g3_w, g3_b, oscale, outp);
}
